// FeatureCommunicationModule_67843303408325
// MI455X (gfx1250) — matrix operand order repaired and checked by reading, not compiled, not run on hardware
//
#include <hip/hip_runtime.h>

// ---------------------------------------------------------------------------
// FeatureCommunicationModule for MI455X (gfx1250, wave32, WMMA)
//
// All heavy GEMMs run on v_wmma_f32_16x16x32_f16 (fp32 activations converted
// to f16 in registers via v_cvt_pk_f16_f32, fp32 accumulation). Weights are
// pre-packed once into f16 B-fragment layout so every B fragment is two
// contiguous b128 loads. Tile loops are compile-time constant (templates) so
// accumulators never spill. LayerNorms are fused into WMMA epilogues / A-
// operand loads. Bilinear deformable sampling is a VALU gather kernel.
// ---------------------------------------------------------------------------

typedef __attribute__((ext_vector_type(16))) _Float16 v16h;
typedef __attribute__((ext_vector_type(8)))  float    v8f;

#define WPB 8  // waves per block (256 threads)

enum { EPI_STORE = 0, EPI_LN = 1, EPI_RESID = 2, EPI_RELU = 3 };

// Problem constants
#define BB   2
#define HS   128
#define WS_  128
#define HI   64
#define WI   64
#define DIM  768
#define INT_ 96
#define NP   4
#define DHEAD 48
#define MQ   (BB * HS * WS_)   // 32768
#define MKV  (BB * HI * WI)    // 8192

// ---------------------------------------------------------------------------
// Helpers: convert 8 consecutive fp32 (two float4) into a[base..base+7] f16
// ---------------------------------------------------------------------------
__device__ __forceinline__ void fcm_cvt8(v16h& a, int base, float4 u, float4 v)
{
    a[base + 0] = (_Float16)u.x;  a[base + 1] = (_Float16)u.y;
    a[base + 2] = (_Float16)u.z;  a[base + 3] = (_Float16)u.w;
    a[base + 4] = (_Float16)v.x;  a[base + 5] = (_Float16)v.y;
    a[base + 6] = (_Float16)v.z;  a[base + 7] = (_Float16)v.w;
}

// ---------------------------------------------------------------------------
// Generic wave-tiled WMMA GEMM (all tile loops compile-time):
//   C[M x (NTtot*16)] = epi( A[M x K] * Wpacked + bias )
//  - TPW: column tiles per wave; CG: column groups per 16-row block
//  - Bp: f16 weights packed as [kt][nt][lane 0..31][16 halves]
//  - ALN: apply LayerNorm (astats row stats + ag/ab) to A elements on load
// ---------------------------------------------------------------------------
template<int TPW, int CG, int EPI, int ALN>
__global__ __launch_bounds__(256)
void fcm_wmma_gemm(const float* __restrict__ A, int lda,
                   const _Float16* __restrict__ Bp, int NTtot,
                   const float* __restrict__ bias,
                   float* __restrict__ Co, int ldc,
                   int K,
                   const float* __restrict__ resid, int ldr,
                   const float* __restrict__ lng, const float* __restrict__ lnb,
                   const float* __restrict__ astats,
                   const float* __restrict__ ag, const float* __restrict__ ab)
{
    const int lane     = threadIdx.x & 31;
    const int wave     = threadIdx.x >> 5;
    const int rbPerBlk = WPB / CG;
    const int rowBlk   = blockIdx.x * rbPerBlk + wave / CG;
    const int row0     = rowBlk * 16;
    const int colGroup = wave % CG;
    const int ct0      = colGroup * TPW;

    const int halfId = lane >> 4;   // 0: lanes 0-15, 1: lanes 16-31
    const int l16    = lane & 15;
    const int arow   = row0 + l16;

    v8f acc[TPW];
    #pragma unroll
    for (int t = 0; t < TPW; ++t)
        #pragma unroll
        for (int j = 0; j < 8; ++j) acc[t][j] = 0.0f;

    float amean = 0.0f, arstd = 1.0f;
    if (ALN) { amean = astats[arow * 2]; arstd = astats[arow * 2 + 1]; }

    const float* Arow = A + (size_t)arow * lda;

    for (int k0 = 0; k0 < K; k0 += 32) {
        if (k0 + 32 < K)
            __builtin_prefetch(Arow + k0 + 32, 0, 1);   // global_prefetch_b8

        // ---- A fragment: 16x32 f16 (ISA 7.12.2 layout) ----
        // Lane holds K in [kb1, kb1+8) and [kb2, kb2+8): two contiguous runs.
        const int kb1 = k0 + halfId * 8;
        const int kb2 = k0 + 16 + halfId * 8;
        float4 u0 = *reinterpret_cast<const float4*>(Arow + kb1);
        float4 u1 = *reinterpret_cast<const float4*>(Arow + kb1 + 4);
        float4 v0 = *reinterpret_cast<const float4*>(Arow + kb2);
        float4 v1 = *reinterpret_cast<const float4*>(Arow + kb2 + 4);
        if (ALN) {
            const float4 g0 = *reinterpret_cast<const float4*>(ag + kb1);
            const float4 g1 = *reinterpret_cast<const float4*>(ag + kb1 + 4);
            const float4 g2 = *reinterpret_cast<const float4*>(ag + kb2);
            const float4 g3 = *reinterpret_cast<const float4*>(ag + kb2 + 4);
            const float4 b0 = *reinterpret_cast<const float4*>(ab + kb1);
            const float4 b1 = *reinterpret_cast<const float4*>(ab + kb1 + 4);
            const float4 b2 = *reinterpret_cast<const float4*>(ab + kb2);
            const float4 b3 = *reinterpret_cast<const float4*>(ab + kb2 + 4);
            u0.x = (u0.x - amean) * arstd * g0.x + b0.x;
            u0.y = (u0.y - amean) * arstd * g0.y + b0.y;
            u0.z = (u0.z - amean) * arstd * g0.z + b0.z;
            u0.w = (u0.w - amean) * arstd * g0.w + b0.w;
            u1.x = (u1.x - amean) * arstd * g1.x + b1.x;
            u1.y = (u1.y - amean) * arstd * g1.y + b1.y;
            u1.z = (u1.z - amean) * arstd * g1.z + b1.z;
            u1.w = (u1.w - amean) * arstd * g1.w + b1.w;
            v0.x = (v0.x - amean) * arstd * g2.x + b2.x;
            v0.y = (v0.y - amean) * arstd * g2.y + b2.y;
            v0.z = (v0.z - amean) * arstd * g2.z + b2.z;
            v0.w = (v0.w - amean) * arstd * g2.w + b2.w;
            v1.x = (v1.x - amean) * arstd * g3.x + b3.x;
            v1.y = (v1.y - amean) * arstd * g3.y + b3.y;
            v1.z = (v1.z - amean) * arstd * g3.z + b3.z;
            v1.w = (v1.w - amean) * arstd * g3.w + b3.w;
        }
        v16h a;
        fcm_cvt8(a, 0, u0, u1);
        fcm_cvt8(a, 8, v0, v1);

        // ---- B fragments (pre-packed f16, contiguous per lane) + WMMA ----
        const int kt = k0 >> 5;
        #pragma unroll
        for (int t = 0; t < TPW; ++t) {
            const size_t boff = ((size_t)(kt * NTtot + (ct0 + t)) * 32 + lane) * 16;
            const v16h b = *reinterpret_cast<const v16h*>(Bp + boff);
            acc[t] = __builtin_amdgcn_wmma_f32_16x16x32_f16(
                false, a, false, b, (short)0, acc[t], false, false);
        }
    }

    // ---- bias ----
    #pragma unroll
    for (int t = 0; t < TPW; ++t) {
        const float bv = bias[(ct0 + t) * 16 + l16];
        #pragma unroll
        for (int j = 0; j < 8; ++j) acc[t][j] += bv;
    }

    if (EPI == EPI_LN) {
        // Requires TPW==6, CG==1, N==96: the full row lives in this wave.
        // Row m = VGPR j=m%8 of lane-half m/8; reduce across 16 lanes.
        #pragma unroll
        for (int j = 0; j < 8; ++j) {
            float s = 0.0f, sq = 0.0f;
            #pragma unroll
            for (int t = 0; t < TPW; ++t) { float v = acc[t][j]; s += v; sq += v * v; }
            #pragma unroll
            for (int m = 1; m < 16; m <<= 1) {
                s  += __shfl_xor(s,  m, 32);
                sq += __shfl_xor(sq, m, 32);
            }
            const float mean = s * (1.0f / 96.0f);
            const float var  = sq * (1.0f / 96.0f) - mean * mean;
            const float rstd = rsqrtf(var + 1e-5f);
            const int row = row0 + j + 8 * halfId;
            float* Crow = Co + (size_t)row * ldc;
            #pragma unroll
            for (int t = 0; t < TPW; ++t) {
                const int c = t * 16 + l16;
                Crow[c] = (acc[t][j] - mean) * rstd * lng[c] + lnb[c];
            }
        }
    } else {
        #pragma unroll
        for (int t = 0; t < TPW; ++t) {
            const int c = (ct0 + t) * 16 + l16;
            #pragma unroll
            for (int j = 0; j < 8; ++j) {
                const int row = row0 + j + 8 * halfId;
                float v = acc[t][j];
                if (EPI == EPI_RESID) v += resid[(size_t)row * ldr + c];
                if (EPI == EPI_RELU)  v = fmaxf(v, 0.0f);
                Co[(size_t)row * ldc + c] = v;
            }
        }
    }
}

// ---------------------------------------------------------------------------
// Pack fp32 weight W[K x N] into f16 B-fragment layout:
//   Bp[((kt*NT + nt)*32 + lane)*16 + j] = W[kt*32 + (lane/16)*8 + (j%8) + (j/8)*16, nt*16 + lane%16]
// ---------------------------------------------------------------------------
__global__ void fcm_pack_b(const float* __restrict__ W, int K, int N,
                           _Float16* __restrict__ Bp)
{
    const int idx = blockIdx.x * 256 + threadIdx.x;
    if (idx >= K * N) return;
    const int j    = idx & 15;
    const int lane = (idx >> 4) & 31;
    const int tile = idx >> 9;
    const int NT   = N >> 4;
    const int kt   = tile / NT, nt = tile % NT;
    const int k    = kt * 32 + (lane >> 4) * 8 + (j & 7) + (j >> 3) * 16;
    const int col  = nt * 16 + (lane & 15);
    Bp[idx] = (_Float16)W[(size_t)k * N + col];
}

// ---------------------------------------------------------------------------
// Build zero-padded combined (off_w | aw_w) fp32 [96 x 32] + bias [32]
// ---------------------------------------------------------------------------
__global__ void fcm_pack_offaw(const float* __restrict__ off_w, const float* __restrict__ off_b,
                               const float* __restrict__ aw_w,  const float* __restrict__ aw_b,
                               float* __restrict__ wcomb, float* __restrict__ bcomb)
{
    const int i = blockIdx.x * blockDim.x + threadIdx.x;
    if (i < 96 * 32) {
        const int k = i >> 5, n = i & 31;
        float v = 0.0f;
        if (n < 16)      v = off_w[k * 16 + n];
        else if (n < 24) v = aw_w[k * 8 + (n - 16)];
        wcomb[i] = v;
    }
    if (i < 32) bcomb[i] = (i < 16) ? off_b[i] : ((i < 24) ? aw_b[i - 16] : 0.0f);
}

// ---------------------------------------------------------------------------
// Per-row LN stats (mean, rstd) over N=768
// ---------------------------------------------------------------------------
__global__ __launch_bounds__(256)
void fcm_ln_stats(const float* __restrict__ X, float* __restrict__ stats, int N)
{
    const int row = blockIdx.x;
    const float* xr = X + (size_t)row * N;
    float s = 0.0f, sq = 0.0f;
    for (int i = threadIdx.x; i < N; i += 256) { float v = xr[i]; s += v; sq += v * v; }
    #pragma unroll
    for (int m = 1; m < 32; m <<= 1) { s += __shfl_xor(s, m, 32); sq += __shfl_xor(sq, m, 32); }
    __shared__ float ss[8], sv[8];
    const int wave = threadIdx.x >> 5;
    if ((threadIdx.x & 31) == 0) { ss[wave] = s; sv[wave] = sq; }
    __syncthreads();
    if (threadIdx.x == 0) {
        float S = 0.0f, Q = 0.0f;
        #pragma unroll
        for (int i = 0; i < 8; ++i) { S += ss[i]; Q += sv[i]; }
        const float mean = S / (float)N;
        const float var  = Q / (float)N - mean * mean;
        stats[row * 2]     = mean;
        stats[row * 2 + 1] = rsqrtf(var + 1e-5f);
    }
}

// ---------------------------------------------------------------------------
// Deformable attention sampling (grid_sample bilinear, zeros padding,
// align_corners=False) + softmax over P, weighted accumulation.
// One block per (b, query); one thread per output channel (head*48 + d).
// offaw rows have stride 32 (cols 0..15 offsets, 16..23 attn logits).
// ---------------------------------------------------------------------------
__device__ __forceinline__ float fcm_vsamp(const float* __restrict__ vb, int x, int y, int c)
{
    if (x < 0 || x >= WI || y < 0 || y >= HI) return 0.0f;
    return vb[(size_t)(y * WI + x) * INT_ + c];
}

__global__ void fcm_deform_sample(const float* __restrict__ offaw,   // [MQ, 32]
                                  const float* __restrict__ value,   // [MKV, 96]
                                  float* __restrict__ opre)          // [MQ, 96]
{
    const int rowIdx = blockIdx.x;            // b*16384 + qi
    const int b  = rowIdx >> 14;
    const int qi = rowIdx & 16383;
    const int t  = threadIdx.x;               // 0..95
    const int head = t / DHEAD;
    const float* oa = offaw + (size_t)rowIdx * 32;

    float lg[NP];
    #pragma unroll
    for (int p = 0; p < NP; ++p) lg[p] = oa[16 + head * NP + p];
    float mx = lg[0];
    #pragma unroll
    for (int p = 1; p < NP; ++p) mx = fmaxf(mx, lg[p]);
    float ex[NP], sum = 0.0f;
    #pragma unroll
    for (int p = 0; p < NP; ++p) { ex[p] = __expf(lg[p] - mx); sum += ex[p]; }
    const float inv = 1.0f / sum;

    const float refx = ((float)(qi & (WS_ - 1)) + 0.5f) / (float)WS_;
    const float refy = ((float)(qi >> 7)        + 0.5f) / (float)HS;
    const float* vb = value + (size_t)b * (HI * WI) * INT_;

    float accv = 0.0f;
    #pragma unroll
    for (int p = 0; p < NP; ++p) {
        const float w  = ex[p] * inv;
        const float ox = oa[head * (NP * 2) + p * 2 + 0];
        const float oy = oa[head * (NP * 2) + p * 2 + 1];
        const float px = refx * (float)WI + ox - 0.5f;
        const float py = refy * (float)HI + oy - 0.5f;
        const float x0f = floorf(px), y0f = floorf(py);
        const int   x0  = (int)x0f,   y0  = (int)y0f;
        const float wx1 = px - x0f,   wy1 = py - y0f;
        const float v00 = fcm_vsamp(vb, x0,     y0,     t);
        const float v10 = fcm_vsamp(vb, x0 + 1, y0,     t);
        const float v01 = fcm_vsamp(vb, x0,     y0 + 1, t);
        const float v11 = fcm_vsamp(vb, x0 + 1, y0 + 1, t);
        const float bi = v00 * (1.0f - wx1) * (1.0f - wy1)
                       + v10 * wx1          * (1.0f - wy1)
                       + v01 * (1.0f - wx1) * wy1
                       + v11 * wx1          * wy1;
        accv += w * bi;
    }
    opre[(size_t)rowIdx * INT_ + t] = accv;
}

// ---------------------------------------------------------------------------
// Host-side orchestration
// ---------------------------------------------------------------------------
extern "C" void kernel_launch(void* const* d_in, const int* in_sizes, int n_in,
                              void* d_out, int out_size, void* d_ws, size_t ws_size,
                              hipStream_t stream)
{
    (void)in_sizes; (void)n_in; (void)out_size; (void)ws_size;

    const float* sam   = (const float*)d_in[0];
    const float* ir    = (const float*)d_in[1];
    const float* q_w   = (const float*)d_in[2];
    const float* q_b   = (const float*)d_in[3];
    const float* kv_w  = (const float*)d_in[4];
    const float* kv_b  = (const float*)d_in[5];
    const float* nq_g  = (const float*)d_in[6];
    const float* nq_b  = (const float*)d_in[7];
    const float* nkv_g = (const float*)d_in[8];
    const float* nkv_b = (const float*)d_in[9];
    const float* off_w = (const float*)d_in[10];
    const float* off_b = (const float*)d_in[11];
    const float* aw_w  = (const float*)d_in[12];
    const float* aw_b  = (const float*)d_in[13];
    const float* val_w = (const float*)d_in[14];
    const float* val_b = (const float*)d_in[15];
    const float* out_w = (const float*)d_in[16];
    const float* out_b = (const float*)d_in[17];
    const float* back_w= (const float*)d_in[18];
    const float* back_b= (const float*)d_in[19];
    const float* nf_g  = (const float*)d_in[20];
    const float* nf_b  = (const float*)d_in[21];
    const float* f1_w  = (const float*)d_in[22];
    const float* f1_b  = (const float*)d_in[23];
    const float* f2_w  = (const float*)d_in[24];
    const float* f2_b  = (const float*)d_in[25];

    float* out = (float*)d_out;          // sam2, then final result (in place)
    float* ws  = (float*)d_ws;

    // ---- fp32 workspace (all offsets multiples of 32 floats) ----
    float* qbuf    = ws;                                   // [MQ  x 96]
    float* kvbuf   = qbuf   + (size_t)MQ  * INT_;          // [MKV x 96]
    float* valbuf  = kvbuf  + (size_t)MKV * INT_;          // [MKV x 96]
    float* oabuf   = valbuf + (size_t)MKV * INT_;          // [MQ  x 32]
    float* opre    = oabuf  + (size_t)MQ  * 32;            // [MQ  x 96]
    float* obuf    = opre   + (size_t)MQ  * INT_;          // [MQ  x 96]
    float* stats   = obuf   + (size_t)MQ  * INT_;          // [MQ  x 2]
    float* tbuf    = stats  + (size_t)MQ  * 2;             // [MQ  x 96]
    float* wcomb   = tbuf   + (size_t)MQ  * INT_;          // [96 x 32]
    float* bcomb   = wcomb  + 96 * 32;                     // [32]
    float* hbase_f = bcomb  + 32;                          // f16 packed weights below

    // ---- packed f16 weights (B-fragment layout) ----
    _Float16* hp     = (_Float16*)hbase_f;
    _Float16* qw_p   = hp;                     // 768*96
    _Float16* kvw_p  = qw_p   + DIM * INT_;
    _Float16* valw_p = kvw_p  + DIM * INT_;    // 96*96
    _Float16* oaw_p  = valw_p + INT_ * INT_;   // 96*32
    _Float16* outw_p = oaw_p  + INT_ * 32;     // 96*96
    _Float16* backw_p= outw_p + INT_ * INT_;   // 96*768
    _Float16* f1w_p  = backw_p+ INT_ * DIM;    // 768*96
    _Float16* f2w_p  = f1w_p  + DIM * INT_;    // 96*768

    // ---- weight packing ----
    fcm_pack_offaw<<<12, 256, 0, stream>>>(off_w, off_b, aw_w, aw_b, wcomb, bcomb);
    fcm_pack_b<<<(DIM  * INT_ + 255) / 256, 256, 0, stream>>>(q_w,   DIM,  INT_, qw_p);
    fcm_pack_b<<<(DIM  * INT_ + 255) / 256, 256, 0, stream>>>(kv_w,  DIM,  INT_, kvw_p);
    fcm_pack_b<<<(INT_ * INT_ + 255) / 256, 256, 0, stream>>>(val_w, INT_, INT_, valw_p);
    fcm_pack_b<<<(INT_ * 32   + 255) / 256, 256, 0, stream>>>(wcomb, INT_, 32,   oaw_p);
    fcm_pack_b<<<(INT_ * INT_ + 255) / 256, 256, 0, stream>>>(out_w, INT_, INT_, outw_p);
    fcm_pack_b<<<(INT_ * DIM  + 255) / 256, 256, 0, stream>>>(back_w,INT_, DIM,  backw_p);
    fcm_pack_b<<<(DIM  * INT_ + 255) / 256, 256, 0, stream>>>(f1_w,  DIM,  INT_, f1w_p);
    fcm_pack_b<<<(INT_ * DIM  + 255) / 256, 256, 0, stream>>>(f2_w,  INT_, DIM,  f2w_p);

    // K1: q = LN(sam @ q_w + q_b)            [32768 x 96], K=768
    fcm_wmma_gemm<6, 1, EPI_LN, 0><<<MQ / 128, 256, 0, stream>>>(
        sam, DIM, qw_p, 6, q_b, qbuf, INT_, DIM,
        nullptr, 0, nq_g, nq_b, nullptr, nullptr, nullptr);

    // K2: kv = LN(ir @ kv_w + kv_b)          [8192 x 96], K=768
    fcm_wmma_gemm<6, 1, EPI_LN, 0><<<MKV / 128, 256, 0, stream>>>(
        ir, DIM, kvw_p, 6, kv_b, kvbuf, INT_, DIM,
        nullptr, 0, nkv_g, nkv_b, nullptr, nullptr, nullptr);

    // K3: value = kv @ val_w + val_b         [8192 x 96], K=96
    fcm_wmma_gemm<6, 1, EPI_STORE, 0><<<MKV / 128, 256, 0, stream>>>(
        kvbuf, INT_, valw_p, 6, val_b, valbuf, INT_, INT_,
        nullptr, 0, nullptr, nullptr, nullptr, nullptr, nullptr);

    // K4: offaw = q @ [off_w|aw_w|0] + bias  [32768 x 32], K=96
    fcm_wmma_gemm<2, 1, EPI_STORE, 0><<<MQ / 128, 256, 0, stream>>>(
        qbuf, INT_, oaw_p, 2, bcomb, oabuf, 32, INT_,
        nullptr, 0, nullptr, nullptr, nullptr, nullptr, nullptr);

    // K5: deformable bilinear sampling + softmax-weighted sum -> opre
    fcm_deform_sample<<<MQ, 96, 0, stream>>>(oabuf, valbuf, opre);

    // K6: o = opre @ out_w + out_b + q       [32768 x 96], K=96
    fcm_wmma_gemm<6, 1, EPI_RESID, 0><<<MQ / 128, 256, 0, stream>>>(
        opre, INT_, outw_p, 6, out_b, obuf, INT_, INT_,
        qbuf, INT_, nullptr, nullptr, nullptr, nullptr, nullptr);

    // K7: sam2 = sam + o @ back_w + back_b   [32768 x 768], K=96 -> d_out
    fcm_wmma_gemm<6, 8, EPI_RESID, 0><<<MQ / 16, 256, 0, stream>>>(
        obuf, INT_, backw_p, 48, back_b, out, DIM, INT_,
        sam, DIM, nullptr, nullptr, nullptr, nullptr, nullptr);

    // K8: per-row LN stats of sam2
    fcm_ln_stats<<<MQ, 256, 0, stream>>>(out, stats, DIM);

    // K9: t = relu(LN(sam2) @ f1_w + f1_b)   [32768 x 96], K=768 (LN fused on A)
    fcm_wmma_gemm<6, 1, EPI_RELU, 1><<<MQ / 128, 256, 0, stream>>>(
        out, DIM, f1w_p, 6, f1_b, tbuf, INT_, DIM,
        nullptr, 0, nullptr, nullptr, stats, nf_g, nf_b);

    // K10: out = sam2 + t @ f2_w + f2_b      [32768 x 768], K=96 (in place)
    fcm_wmma_gemm<6, 8, EPI_RESID, 0><<<MQ / 16, 256, 0, stream>>>(
        tbuf, INT_, f2w_p, 48, f2_b, out, DIM, INT_,
        out, DIM, nullptr, nullptr, nullptr, nullptr, nullptr);
}
